// Attn_Conv_Module_39883066310718
// MI455X (gfx1250) — matrix operand order repaired and checked by reading, not compiled, not run on hardware
//
#include <hip/hip_runtime.h>
#include <hip/hip_bf16.h>

typedef __attribute__((ext_vector_type(16))) __bf16 v16bf;
typedef __attribute__((ext_vector_type(8)))  float  v8f;

union U16BF { v16bf v; unsigned u[8]; };

static __device__ __forceinline__ unsigned short f2bf(float f) {
    unsigned u = __float_as_uint(f);
    u += 0x7FFFu + ((u >> 16) & 1u);   // round-to-nearest-even
    return (unsigned short)(u >> 16);
}

// ---------------------------------------------------------------------------
// Tensor Data Mover: build D# groups (CDNA5 ISA) and issue a
// 2D tile load Global -> LDS.  Tracked by TENSORcnt.
// ---------------------------------------------------------------------------
#if __has_builtin(__builtin_amdgcn_tensor_load_to_lds)
#define HAVE_TDM 1
typedef __attribute__((ext_vector_type(4))) unsigned tdm_g0_t;
typedef __attribute__((ext_vector_type(8))) int      tdm_g1_t;
typedef __attribute__((ext_vector_type(4))) int      tdm_g2_t;

static __device__ __forceinline__ void tdm_load_2d(
        unsigned lds_off,            // LDS byte offset (low 32 of generic ptr)
        const void* gaddr,           // global tile start
        unsigned ds_code,            // 0=1B 1=2B 2=4B 3=8B
        unsigned td0, unsigned td1,  // tensor dims (data_size units)
        unsigned t0, unsigned t1,    // tile dims
        unsigned long long stride0)  // tensor_dim0_stride (data_size units)
{
    unsigned long long ga = (unsigned long long)(uintptr_t)gaddr;
    tdm_g0_t g0;
    g0[0] = 1u;                                           // count=1, user desc
    g0[1] = lds_off;                                      // lds_addr [63:32]
    g0[2] = (unsigned)(ga & 0xFFFFFFFFull);               // global_addr[31:0]
    g0[3] = (unsigned)((ga >> 32) & 0x1FFFFFFull)         // global_addr[56:32]
          | (2u << 30);                                   // type=2 ("image")
    tdm_g1_t g1;
    g1[0] = (int)(ds_code << 16);                         // data_size; mask=0
    g1[1] = (int)((td0 & 0xFFFFu) << 16);                 // tensor_dim0[15:0]
    g1[2] = (int)(((td0 >> 16) & 0xFFFFu) | ((td1 & 0xFFFFu) << 16));
    g1[3] = (int)(((td1 >> 16) & 0xFFFFu) | ((t0 & 0xFFFFu) << 16));
    g1[4] = (int)(t1 & 0xFFFFu);                          // tile_dim1; dim2=0
    g1[5] = (int)(stride0 & 0xFFFFFFFFull);               // dim0_stride[31:0]
    g1[6] = (int)((stride0 >> 32) & 0xFFFFull);           // dim0_stride[47:32]
    g1[7] = 0;                                            // dim1_stride unused
    tdm_g2_t gz = {0, 0, 0, 0};
#if defined(__clang_major__) && __clang_major__ >= 23
    tdm_g1_t gz8 = {0, 0, 0, 0, 0, 0, 0, 0};
    __builtin_amdgcn_tensor_load_to_lds(g0, g1, gz, gz, gz8, 0);
#else
    __builtin_amdgcn_tensor_load_to_lds(g0, g1, gz, gz, 0);
#endif
}
#else
#define HAVE_TDM 0
#endif

// ---------------------------------------------------------------------------
// fp32 -> bf16 flat convert (weights)
// ---------------------------------------------------------------------------
__global__ void k_convert(const float* __restrict__ src,
                          unsigned short* __restrict__ dst, int n) {
    int i = blockIdx.x * blockDim.x + threadIdx.x;
    if (i < n) dst[i] = f2bf(src[i]);
}

// ---------------------------------------------------------------------------
// x [B][C][L] fp32 -> xT [B][L][C] bf16  (so GEMM B-operand k-pairs contiguous)
// ---------------------------------------------------------------------------
__global__ void k_transpose_x(const float* __restrict__ x,
                              unsigned short* __restrict__ xT,
                              int C, int L) {
    int b = blockIdx.y;
    long e = (long)blockIdx.x * blockDim.x + threadIdx.x;   // e in [0, C*L)
    long CL = (long)C * L;
    if (e >= CL) return;
    int ch  = (int)(e / L);
    int pos = (int)(e % L);
    float v = x[(long)b * CL + e];
    xT[(long)b * CL + (long)pos * C + ch] = f2bf(v);
}

// ---------------------------------------------------------------------------
// Generic bf16 WMMA GEMM:  D(MxN) = A(MxK) * B(KxN) (+bias) (*scale)
//   A  : bf16 row-major [M][K]            (weights, shared across batch)
//   BT : bf16 [N][K] per batch            (activation, transposed storage)
// Outputs (any may be null):
//   outT : bf16 [N][M]   per batch
//   outR : bf16 [M][N]   per batch
//   outF : fp32 [rowOff+M][ldF] per batch
// One wave computes a 16x64 tile: A regs reused across 4 N-tiles.
// ---------------------------------------------------------------------------
__global__ void k_gemm(const unsigned short* __restrict__ A,
                       const unsigned short* __restrict__ BT,
                       const float* __restrict__ bias,
                       int M, int K, int N, long strideBT,
                       unsigned short* __restrict__ outT, long strideOT,
                       unsigned short* __restrict__ outR, long strideOR,
                       float* __restrict__ outF, long strideOF, int ldF, int rowOffF,
                       const float* __restrict__ scale_ptr) {
    const int lane = threadIdx.x;
    const int nl = lane & 15;
    const int h  = lane >> 4;
    const int n0 = blockIdx.x * 64;
    const int m0 = blockIdx.y * 16;
    const int b  = blockIdx.z;
    const unsigned short* Bb = BT + (long)b * strideBT;

    v8f zero = {};
    v8f acc[4];
    acc[0] = zero; acc[1] = zero; acc[2] = zero; acc[3] = zero;

    for (int k0 = 0; k0 < K; k0 += 32) {
        U16BF a;
        #pragma unroll
        for (int i = 0; i < 8; ++i) {
            int kk = k0 + h * 16 + 2 * i;
            a.u[i] = *(const unsigned*)(A + (long)(m0 + nl) * K + kk);
        }
        #pragma unroll
        for (int t = 0; t < 4; ++t) {
            U16BF bm;
            int n = n0 + t * 16 + nl;
            #pragma unroll
            for (int i = 0; i < 8; ++i) {
                int kk = k0 + h * 16 + 2 * i;
                bm.u[i] = *(const unsigned*)(Bb + (long)n * K + kk);
            }
            acc[t] = __builtin_amdgcn_wmma_f32_16x16x32_bf16(
                false, a.v, false, bm.v, (short)0, acc[t], false, false);
        }
    }

    float sc = scale_ptr ? *scale_ptr : 1.0f;
    #pragma unroll
    for (int t = 0; t < 4; ++t) {
        #pragma unroll
        for (int i = 0; i < 8; ++i) {
            int m = m0 + i + 8 * h;
            int n = n0 + t * 16 + nl;
            float val = (acc[t][i] + (bias ? bias[m] : 0.0f)) * sc;
            if (outF) outF[(long)b * strideOF + (long)(rowOffF + m) * ldF + n] = val;
            unsigned short hv = f2bf(val);
            if (outT) outT[(long)b * strideOT + (long)n * M + m] = hv;
            if (outR) outR[(long)b * strideOR + (long)m * N + n] = hv;
        }
    }
}

// ---------------------------------------------------------------------------
// Fused flash attention:  O[i,:] = softmax_j(q[:,i].k[:,j]) * v[:,j]^T
//   qT/kT : bf16 [B][L][c], vb : bf16 [B][c][L], oT : bf16 [B][L][c]
// One wave per (batch, 16-row query block). Q resident in A-layout VGPRs,
// 16x256 fp32 accumulator resident, online softmax across 32-wide j-tiles.
// K/V tiles are staged into double-buffered LDS by the Tensor Data Mover
// (tensor_load_to_lds + s_wait_tensorcnt) so DMA of block j+1 overlaps the
// 32 WMMAs of block j.
// ---------------------------------------------------------------------------
__global__ void k_flash(const unsigned short* __restrict__ qT,
                        const unsigned short* __restrict__ kT,
                        const unsigned short* __restrict__ vb,
                        unsigned short* __restrict__ oT,
                        int L, int Cc) {
    const int lane = threadIdx.x;
    const int nl = lane & 15;
    const int h  = lane >> 4;
    const int i0 = blockIdx.x * 16;
    const int b  = blockIdx.y;
    const long actOff = (long)b * L * Cc;
    const unsigned short* q = qT + actOff;
    const unsigned short* k = kT + actOff;
    const unsigned short* v = vb + (long)b * Cc * L;

    __shared__ __align__(16) unsigned short pb[16 * 32];
#if HAVE_TDM
    __shared__ __align__(16) unsigned short kbuf[2][32 * 256];  // [j][c] 16KB
    __shared__ __align__(16) unsigned short vbuf[2][256 * 32];  // [c][j] 16KB
#endif

    // Resident Q block, A-layout, 8 K-chunks of 32
    U16BF aq[8];
    #pragma unroll
    for (int ch = 0; ch < 8; ++ch) {
        #pragma unroll
        for (int i = 0; i < 8; ++i) {
            int kk = ch * 32 + h * 16 + 2 * i;
            aq[ch].u[i] = *(const unsigned*)(q + (long)(i0 + nl) * Cc + kk);
        }
    }

    v8f zero = {};
    v8f o[16];
    #pragma unroll
    for (int t = 0; t < 16; ++t) o[t] = zero;
    float mrun[8], lrun[8];
    #pragma unroll
    for (int r = 0; r < 8; ++r) { mrun[r] = -1e30f; lrun[r] = 0.0f; }

#if HAVE_TDM
    // Prologue: DMA block 0 into buffer 0.
    // K-tile: 32 contiguous rows of kT = 16KB flat copy (8B elements).
    // V-tile: 256 rows x 32 bf16, row stride L -> packed [256][32] in LDS.
    tdm_load_2d((unsigned)(uintptr_t)&kbuf[0][0], k, 3,
                2048, 1, 2048, 1, 2048ull);
    tdm_load_2d((unsigned)(uintptr_t)&vbuf[0][0], v, 1,
                32, 256, 32, 256, (unsigned long long)L);
    int cur = 0;
#endif

    for (int j0 = 0; j0 < L; j0 += 32) {
#if HAVE_TDM
        if (j0 + 32 < L) {
            // Issue DMA for the next j-block into the other buffer, then wait
            // only for the current block's two loads (TDM in-order per wave).
            tdm_load_2d((unsigned)(uintptr_t)&kbuf[cur ^ 1][0],
                        k + (long)(j0 + 32) * Cc, 3, 2048, 1, 2048, 1, 2048ull);
            tdm_load_2d((unsigned)(uintptr_t)&vbuf[cur ^ 1][0],
                        v + (j0 + 32), 1, 32, 256, 32, 256,
                        (unsigned long long)L);
            __builtin_amdgcn_s_wait_tensorcnt(2);
        } else {
            __builtin_amdgcn_s_wait_tensorcnt(0);
        }
        const unsigned short* ksrc = &kbuf[cur][0];     // [jr][256]
        const unsigned short* vsrc = &vbuf[cur][0];     // [ch][32]
        const int vld = 32, voff = 0;
#else
        const unsigned short* ksrc = k + (long)j0 * Cc;
        const unsigned short* vsrc = v;
        const int vld = L, voff = j0;
#endif
        // E = Q * K^T for 32 j-columns (two 16x16 tiles)
        v8f e0 = zero, e1 = zero;
        #pragma unroll
        for (int ch = 0; ch < 8; ++ch) {
            U16BF b0, b1;
            #pragma unroll
            for (int i = 0; i < 8; ++i) {
                int kk = ch * 32 + h * 16 + 2 * i;
                b0.u[i] = *(const unsigned*)(ksrc + (long)nl * Cc + kk);
                b1.u[i] = *(const unsigned*)(ksrc + (long)(16 + nl) * Cc + kk);
            }
            e0 = __builtin_amdgcn_wmma_f32_16x16x32_bf16(
                false, aq[ch].v, false, b0.v, (short)0, e0, false, false);
            e1 = __builtin_amdgcn_wmma_f32_16x16x32_bf16(
                false, aq[ch].v, false, b1.v, (short)0, e1, false, false);
        }

        // Online softmax per output row r (row m = r + 8h lives in one 16-lane half)
        float p0[8], p1[8], alpha[8];
        #pragma unroll
        for (int r = 0; r < 8; ++r) {
            float mx = fmaxf(e0[r], e1[r]);
            #pragma unroll
            for (int d = 1; d < 16; d <<= 1) mx = fmaxf(mx, __shfl_xor(mx, d, 32));
            float nm = fmaxf(mrun[r], mx);
            alpha[r] = __expf(mrun[r] - nm);
            p0[r] = __expf(e0[r] - nm);
            p1[r] = __expf(e1[r] - nm);
            float rs = p0[r] + p1[r];
            #pragma unroll
            for (int d = 1; d < 16; d <<= 1) rs += __shfl_xor(rs, d, 32);
            lrun[r] = lrun[r] * alpha[r] + rs;
            mrun[r] = nm;
        }

        // P (16x32) -> LDS row-major, then reload in WMMA A-layout bf16
        #pragma unroll
        for (int r = 0; r < 8; ++r) {
            pb[(r + 8 * h) * 32 + nl]      = f2bf(p0[r]);
            pb[(r + 8 * h) * 32 + 16 + nl] = f2bf(p1[r]);
        }
        asm volatile("s_wait_dscnt 0" ::: "memory");
        U16BF pa;
        #pragma unroll
        for (int i = 0; i < 8; ++i) {
            int kk = h * 16 + 2 * i;
            pa.u[i] = *(const unsigned*)(pb + nl * 32 + kk);
        }
        asm volatile("s_wait_dscnt 0" ::: "memory");

        // Rescale running accumulator, then O += P * V^T
        #pragma unroll
        for (int t = 0; t < 16; ++t) {
            #pragma unroll
            for (int r = 0; r < 8; ++r) o[t][r] *= alpha[r];
        }
        #pragma unroll
        for (int t = 0; t < 16; ++t) {
            U16BF bvv;
            #pragma unroll
            for (int i = 0; i < 8; ++i) {
                int jj = voff + h * 16 + 2 * i;
                bvv.u[i] = *(const unsigned*)(vsrc + (long)(t * 16 + nl) * vld + jj);
            }
            o[t] = __builtin_amdgcn_wmma_f32_16x16x32_bf16(
                false, pa.v, false, bvv.v, (short)0, o[t], false, false);
        }
#if HAVE_TDM
        cur ^= 1;
#endif
    }

    // Normalize and store transposed [L][c] bf16
    float inv[8];
    #pragma unroll
    for (int r = 0; r < 8; ++r) inv[r] = 1.0f / lrun[r];
    #pragma unroll
    for (int t = 0; t < 16; ++t) {
        #pragma unroll
        for (int r = 0; r < 8; ++r) {
            int irow = i0 + r + 8 * h;
            int n = t * 16 + nl;
            oT[actOff + (long)irow * Cc + n] = f2bf(o[t][r] * inv[r]);
        }
    }
}

// ---------------------------------------------------------------------------
extern "C" void kernel_launch(void* const* d_in, const int* in_sizes, int n_in,
                              void* d_out, int out_size, void* d_ws, size_t ws_size,
                              hipStream_t stream) {
    const int B = 8, C = 512, L = 2048, c = 256;
    const float* x  = (const float*)d_in[0];
    const float* Wv = (const float*)d_in[1];
    const float* bv = (const float*)d_in[2];
    const float* Wq = (const float*)d_in[3];
    const float* bq = (const float*)d_in[4];
    const float* Wk = (const float*)d_in[5];
    const float* bk = (const float*)d_in[6];
    const float* Wc = (const float*)d_in[7];
    const float* bc = (const float*)d_in[8];
    const float* gamma = (const float*)d_in[9];
    float* out = (float*)d_out;

    // Workspace carve-out
    char* w = (char*)d_ws;
    auto alloc = [&](size_t bytes) {
        char* p = w;
        w += (bytes + 255) & ~(size_t)255;
        return p;
    };
    unsigned short* xT  = (unsigned short*)alloc((size_t)B * L * C * 2);  // [B][L][C]
    unsigned short* WvH = (unsigned short*)alloc((size_t)c * C * 2);
    unsigned short* WqH = (unsigned short*)alloc((size_t)c * c * 2);
    unsigned short* WkH = (unsigned short*)alloc((size_t)c * c * 2);
    unsigned short* WcH = (unsigned short*)alloc((size_t)c * c * 2);
    unsigned short* vT  = (unsigned short*)alloc((size_t)B * L * c * 2);  // [B][L][c]
    unsigned short* vR  = (unsigned short*)alloc((size_t)B * c * L * 2);  // [B][c][L]
    unsigned short* qT  = (unsigned short*)alloc((size_t)B * L * c * 2);
    unsigned short* kTw = (unsigned short*)alloc((size_t)B * L * c * 2);
    unsigned short* oT  = (unsigned short*)alloc((size_t)B * L * c * 2);

    // 0) weights fp32 -> bf16
    k_convert<<<dim3((c * C + 255) / 256), 256, 0, stream>>>(Wv, WvH, c * C);
    k_convert<<<dim3((c * c + 255) / 256), 256, 0, stream>>>(Wq, WqH, c * c);
    k_convert<<<dim3((c * c + 255) / 256), 256, 0, stream>>>(Wk, WkH, c * c);
    k_convert<<<dim3((c * c + 255) / 256), 256, 0, stream>>>(Wc, WcH, c * c);

    // 1) x -> xT bf16
    k_transpose_x<<<dim3((C * L + 255) / 256, B), 256, 0, stream>>>(x, xT, C, L);

    // 2) v = Wv x + bv : fp32 straight into out[:, 0:c, :], bf16 in both layouts
    k_gemm<<<dim3(L / 64, c / 16, B), 32, 0, stream>>>(
        WvH, xT, bv, c, C, L, (long)L * C,
        vT, (long)L * c, vR, (long)c * L,
        out, (long)2 * c * L, L, 0, nullptr);

    // 3) q = Wq v + bq ; k = Wk v + bk  (transposed bf16 outputs)
    k_gemm<<<dim3(L / 64, c / 16, B), 32, 0, stream>>>(
        WqH, vT, bq, c, c, L, (long)L * c,
        qT, (long)L * c, nullptr, 0, nullptr, 0, 0, 0, nullptr);
    k_gemm<<<dim3(L / 64, c / 16, B), 32, 0, stream>>>(
        WkH, vT, bk, c, c, L, (long)L * c,
        kTw, (long)L * c, nullptr, 0, nullptr, 0, 0, 0, nullptr);

    // 4) fused energy/softmax/attn*v with TDM-staged K/V tiles
    k_flash<<<dim3(L / 16, B), 32, 0, stream>>>(qT, kTw, vR, oT, L, c);

    // 5) out[:, c:2c, :] = gamma * (Wc attn_out + bc)
    k_gemm<<<dim3(L / 64, c / 16, B), 32, 0, stream>>>(
        WcH, oT, bc, c, c, L, (long)L * c,
        nullptr, 0, nullptr, 0,
        out, (long)2 * c * L, L, c, gamma);
}
